// GeometricTransformerModule_80058190397963
// MI455X (gfx1250) — matrix operand order repaired and checked by reading, not compiled, not run on hardware
//
#include <hip/hip_runtime.h>

// MI455X / gfx1250, wave32. bf16 WMMA (16x16x32, f32 accum) for all GEMMs.
// Edge structure exploited: dst = repeat(arange(N), KNN) -> no atomics.
// v3: padded workspace activations -> unguarded stores/loads (no exec ladders),
//     stage-level full/partial split for d_out, dual-accumulator WMMA chains
//     so the scheduler keeps 8+ b128 loads in flight per wave.

typedef __attribute__((ext_vector_type(16))) __bf16  bf16x16;
typedef __attribute__((ext_vector_type(8)))  float   floatx8;
typedef __attribute__((ext_vector_type(4)))  float   floatx4;
typedef __attribute__((ext_vector_type(4)))  unsigned int uint32x4;
typedef __attribute__((ext_vector_type(2)))  unsigned int uint32x2;

union Frag16 { bf16x16 v; uint32x4 q[2]; };

__device__ __forceinline__ unsigned short f2bf(float x) {
    unsigned int u = __float_as_uint(x);
    u += 0x7FFFu + ((u >> 16) & 1u);          // round-to-nearest-even
    return (unsigned short)(u >> 16);
}
__device__ __forceinline__ float bf2f(unsigned short h) {
    return __uint_as_float(((unsigned int)h) << 16);
}

// A fragment: 16x32 bf16 tile, ISA 7.12.2 layout.
// lanes 0-15: K = k0+0..7, k0+16..23 ; lanes 16-31: K = k0+8..15, k0+24..31
__device__ __forceinline__ bf16x16 load_afrag(const unsigned short* M, size_t stride,
                                              int row, int k0, int lane) {
    int kofs = k0 + (((lane >> 4) & 1) << 3);
    const unsigned short* p = M + (size_t)row * stride + kofs;
    Frag16 f;
    f.q[0] = *(const uint32x4*)p;
    f.q[1] = *(const uint32x4*)(p + 16);
    return f.v;
}

// B fragment: 32x16 tile from transposed weight Wt[col][k].
// lanes 0-15: K = k0+0..7, k0+16..23 ; lanes 16-31: K = k0+8..15, k0+24..31 ; lane&15 = column.
__device__ __forceinline__ bf16x16 load_bfrag(const unsigned short* Wt, int Kdim,
                                              int k0, int n0, int lane) {
    int col  = n0 + (lane & 15);
    int kofs = k0 + (((lane >> 4) & 1) << 3);
    const unsigned short* p = Wt + (size_t)col * Kdim + kofs;
    Frag16 f;
    f.q[0] = *(const uint32x4*)p;
    f.q[1] = *(const uint32x4*)(p + 16);
    return f.v;
}

__device__ __forceinline__ floatx8 wmma_bf16(bf16x16 a, bf16x16 b, floatx8 c) {
    return __builtin_amdgcn_wmma_f32_16x16x32_bf16(false, a, false, b, (short)0, c,
                                                   false, false);
}

__device__ __forceinline__ floatx8 zero8() {
    floatx8 z;
#pragma unroll
    for (int i = 0; i < 8; ++i) z[i] = 0.0f;
    return z;
}

// ---------------- weight convert+transpose: W[rows][cols] f32 -> Wt[cols][rows] bf16
__global__ void wt_cvt_kernel(const float* __restrict__ W, unsigned short* __restrict__ Wt,
                              int rows, int cols) {
    int idx = blockIdx.x * blockDim.x + threadIdx.x;
    if (idx >= rows * cols) return;
    int r = idx / cols, c = idx - r * cols;
    Wt[(size_t)c * rows + r] = f2bf(W[idx]);
}

// ---------------- kernel 1: h = BN1(f); q,k,v = h @ W{q,k,v}  (stored bf16, padded rows)
__global__ __launch_bounds__(256) void qkv_kernel(
    const float* __restrict__ f,
    const float* __restrict__ g1, const float* __restrict__ b1,
    const float* __restrict__ m1, const float* __restrict__ v1,
    const unsigned short* __restrict__ Wqt, const unsigned short* __restrict__ Wkt,
    const unsigned short* __restrict__ Wvt,
    unsigned short* __restrict__ qb, unsigned short* __restrict__ kb,
    unsigned short* __restrict__ vb, int n)
{
    __shared__ unsigned short hs[128 * 136];   // bf16 BN1(f) tile, padded rows
    const int tid  = threadIdx.x;
    const int base = blockIdx.x * 128;

    // stage BN1(f) tile into LDS as bf16 (clamped read, no divergent ladder)
    for (int i = tid; i < 4096; i += 256) {     // 4096 float4 = 128x128
        int row = i >> 5;
        int c4  = (i & 31) << 2;
        int grow = min(base + row, n - 1);
        floatx4 fv = *(const floatx4*)(f + (size_t)grow * 128 + c4);
        floatx4 gv = *(const floatx4*)(g1 + c4);
        floatx4 bv = *(const floatx4*)(b1 + c4);
        floatx4 mv = *(const floatx4*)(m1 + c4);
        floatx4 vv = *(const floatx4*)(v1 + c4);
        unsigned short* o = hs + row * 136 + c4;
#pragma unroll
        for (int t = 0; t < 4; ++t) {
            float s = gv[t] * rsqrtf(vv[t] + 1e-5f);
            o[t] = f2bf((fv[t] - mv[t]) * s + bv[t]);
        }
    }
    __syncthreads();

    const int lane = tid & 31, w = tid >> 5;
    const int m0 = w * 16;
    bf16x16 a[4];
#pragma unroll
    for (int kk = 0; kk < 4; ++kk)
        a[kk] = load_afrag(hs, 136, m0 + (lane & 15), kk * 32, lane);

    const int col   = lane & 15;
    const int rbase = m0 + (((lane >> 4) & 1) << 3);
#pragma unroll
    for (int m = 0; m < 3; ++m) {
        const unsigned short* Wt = (m == 0) ? Wqt : ((m == 1) ? Wkt : Wvt);
        unsigned short* Ob       = (m == 0) ? qb  : ((m == 1) ? kb  : vb);
        for (int n0 = 0; n0 < 128; n0 += 32) {          // two 16-col tiles per step
            bf16x16 bA[4], bB[4];
#pragma unroll
            for (int kk = 0; kk < 4; ++kk) {
                bA[kk] = load_bfrag(Wt, 128, kk * 32, n0,      lane);
                bB[kk] = load_bfrag(Wt, 128, kk * 32, n0 + 16, lane);
            }
            floatx8 accA = zero8(), accB = zero8();
#pragma unroll
            for (int kk = 0; kk < 4; ++kk) {
                accA = wmma_bf16(a[kk], bA[kk], accA);  // two independent chains
                accB = wmma_bf16(a[kk], bB[kk], accB);
            }
            unsigned short* op = Ob + (size_t)(base + rbase) * 128 + n0 + col;
#pragma unroll
            for (int r = 0; r < 8; ++r) {               // padded buffer: no guards
                op[(size_t)r * 128]      = f2bf(accA[r]);
                op[(size_t)r * 128 + 16] = f2bf(accB[r]);
            }
        }
    }
}

// ---------------- kernel 2: edge attention. One wave per node (edges sorted by dst, knn each).
__global__ __launch_bounds__(256) void attn_kernel(
    const unsigned short* __restrict__ qb, const unsigned short* __restrict__ kb,
    const unsigned short* __restrict__ vb, const int* __restrict__ src,
    unsigned short* __restrict__ ab, int n, int knn)
{
    const int wid  = (blockIdx.x * blockDim.x + threadIdx.x) >> 5;
    const int lane = threadIdx.x & 31;
    if (wid >= n) return;
    const int c0 = lane << 2;                      // 4 channels per lane; head = lane>>3

    uint32x2 qraw = *(const uint32x2*)(qb + (size_t)wid * 128 + c0);
    float qv[4] = { bf2f((unsigned short)(qraw[0] & 0xFFFF)),
                    bf2f((unsigned short)(qraw[0] >> 16)),
                    bf2f((unsigned short)(qraw[1] & 0xFFFF)),
                    bf2f((unsigned short)(qraw[1] >> 16)) };
    float wv[4] = {0.f, 0.f, 0.f, 0.f};
    float z = 0.f;
    const size_t ebase = (size_t)wid * knn;

    for (int j = 0; j < knn; ++j) {
        int s = src[ebase + j];
        uint32x2 kr = *(const uint32x2*)(kb + (size_t)s * 128 + c0);
        float p = qv[0] * bf2f((unsigned short)(kr[0] & 0xFFFF))
                + qv[1] * bf2f((unsigned short)(kr[0] >> 16))
                + qv[2] * bf2f((unsigned short)(kr[1] & 0xFFFF))
                + qv[3] * bf2f((unsigned short)(kr[1] >> 16));
        p += __shfl_xor(p, 1, 32);                 // reduce within 8-lane head group
        p += __shfl_xor(p, 2, 32);
        p += __shfl_xor(p, 4, 32);
        float sc = p * 0.17677669529663687f;       // 1/sqrt(32)
        sc = fminf(5.f, fmaxf(-5.f, sc));
        float wt = __expf(sc);
        z += wt;
        uint32x2 vr = *(const uint32x2*)(vb + (size_t)s * 128 + c0);
        wv[0] += wt * bf2f((unsigned short)(vr[0] & 0xFFFF));
        wv[1] += wt * bf2f((unsigned short)(vr[0] >> 16));
        wv[2] += wt * bf2f((unsigned short)(vr[1] & 0xFFFF));
        wv[3] += wt * bf2f((unsigned short)(vr[1] >> 16));
    }
    float inv = 1.f / (z + 1e-6f);
    unsigned short* o = ab + (size_t)wid * 128 + c0;
#pragma unroll
    for (int t = 0; t < 4; ++t) o[t] = f2bf(wv[t] * inv);
}

// ---------------- kernel 3: node = attn@Wo + bo + f ; h2 = BN2(node) ;
//                  t = silu(h2@W1) ; out = node + t@W2     (all GEMMs via WMMA)
__global__ __launch_bounds__(256) void fused_out_kernel(
    const unsigned short* __restrict__ ab, const unsigned short* __restrict__ Wot,
    const float* __restrict__ bo, const float* __restrict__ f,
    const float* __restrict__ g2, const float* __restrict__ b2,
    const float* __restrict__ m2, const float* __restrict__ v2,
    const unsigned short* __restrict__ W1t, const unsigned short* __restrict__ W2t,
    float* __restrict__ out, int n)
{
    __shared__ unsigned short h2s[128 * 136];   // bf16 BN2(node)
    __shared__ unsigned short ts [128 * 264];   // bf16 silu(h2@W1), 256 cols + pad
    const int tid = threadIdx.x, lane = tid & 31, w = tid >> 5;
    const int base = blockIdx.x * 128, m0 = w * 16;
    const int col = lane & 15;
    const int rbase = m0 + (((lane >> 4) & 1) << 3);
    const bool full = (base + 128) <= n;        // uniform per block

    // ---- stage A: node = attn@Wo + bo + f -> out (fp32) & h2s (bf16 BN2)
    {
        bf16x16 a[4];                           // ab is padded: unguarded loads
#pragma unroll
        for (int kk = 0; kk < 4; ++kk)
            a[kk] = load_afrag(ab, 128, base + m0 + (lane & 15), kk * 32, lane);
        if (full) {
            for (int n0 = 0; n0 < 128; n0 += 32) {
                bf16x16 bA[4], bB[4];
#pragma unroll
                for (int kk = 0; kk < 4; ++kk) {
                    bA[kk] = load_bfrag(Wot, 128, kk * 32, n0,      lane);
                    bB[kk] = load_bfrag(Wot, 128, kk * 32, n0 + 16, lane);
                }
                floatx8 accA = zero8(), accB = zero8();
#pragma unroll
                for (int kk = 0; kk < 4; ++kk) {
                    accA = wmma_bf16(a[kk], bA[kk], accA);
                    accB = wmma_bf16(a[kk], bB[kk], accB);
                }
#pragma unroll
                for (int half = 0; half < 2; ++half) {
                    const floatx8& acc = half ? accB : accA;
                    int cg = n0 + half * 16 + col;
                    float boc = bo[cg];
                    float sc2 = g2[cg] * rsqrtf(v2[cg] + 1e-5f);
                    float mb = m2[cg], bb = b2[cg];
#pragma unroll
                    for (int r = 0; r < 8; ++r) {
                        int row = rbase + r;
                        size_t gi = (size_t)(base + row) * 128 + cg;
                        float node = acc[r] + boc + f[gi];
                        out[gi] = node;          // re-read by SAME thread in stage C
                        h2s[row * 136 + cg] = f2bf((node - mb) * sc2 + bb);
                    }
                }
            }
        } else {                                 // cold path: boundary block only
            for (int n0 = 0; n0 < 128; n0 += 16) {
                bf16x16 bfr[4];
#pragma unroll
                for (int kk = 0; kk < 4; ++kk)
                    bfr[kk] = load_bfrag(Wot, 128, kk * 32, n0, lane);
                floatx8 acc = zero8();
#pragma unroll
                for (int kk = 0; kk < 4; ++kk)
                    acc = wmma_bf16(a[kk], bfr[kk], acc);
                int cg = n0 + col;
                float boc = bo[cg];
                float sc2 = g2[cg] * rsqrtf(v2[cg] + 1e-5f);
                float mb = m2[cg], bb = b2[cg];
#pragma unroll
                for (int r = 0; r < 8; ++r) {
                    int row = rbase + r, grow = base + row;
                    int grc = min(grow, n - 1);
                    float node = acc[r] + boc + f[(size_t)grc * 128 + cg];
                    if (grow < n) out[(size_t)grow * 128 + cg] = node;
                    h2s[row * 136 + cg] = f2bf((node - mb) * sc2 + bb);
                }
            }
        }
    }
    __syncthreads();

    // ---- stage B: t = silu(h2 @ W1)  (128 -> 256), no guards anywhere
    {
        bf16x16 a[4];
#pragma unroll
        for (int kk = 0; kk < 4; ++kk)
            a[kk] = load_afrag(h2s, 136, m0 + (lane & 15), kk * 32, lane);
        for (int n0 = 0; n0 < 256; n0 += 32) {
            bf16x16 bA[4], bB[4];
#pragma unroll
            for (int kk = 0; kk < 4; ++kk) {
                bA[kk] = load_bfrag(W1t, 128, kk * 32, n0,      lane);
                bB[kk] = load_bfrag(W1t, 128, kk * 32, n0 + 16, lane);
            }
            floatx8 accA = zero8(), accB = zero8();
#pragma unroll
            for (int kk = 0; kk < 4; ++kk) {
                accA = wmma_bf16(a[kk], bA[kk], accA);
                accB = wmma_bf16(a[kk], bB[kk], accB);
            }
#pragma unroll
            for (int r = 0; r < 8; ++r) {
                float xa = accA[r], xb = accB[r];
                float sa = 1.f / (1.f + __expf(-xa));
                float sb = 1.f / (1.f + __expf(-xb));
                unsigned short* tp = ts + (rbase + r) * 264 + n0 + col;
                tp[0]  = f2bf(xa * sa);
                tp[16] = f2bf(xb * sb);
            }
        }
    }
    __syncthreads();

    // ---- stage C: out = node + t @ W2  (256 -> 128)
    {
        bf16x16 a8[8];
#pragma unroll
        for (int kk = 0; kk < 8; ++kk)
            a8[kk] = load_afrag(ts, 264, m0 + (lane & 15), kk * 32, lane);
        if (full) {
            for (int n0 = 0; n0 < 128; n0 += 16) {
                bf16x16 bfr[8];
#pragma unroll
                for (int kk = 0; kk < 8; ++kk)
                    bfr[kk] = load_bfrag(W2t, 256, kk * 32, n0, lane);
                floatx8 acc = zero8();
#pragma unroll
                for (int kk = 0; kk < 8; ++kk)
                    acc = wmma_bf16(a8[kk], bfr[kk], acc);
                float* op = out + (size_t)(base + rbase) * 128 + n0 + col;
#pragma unroll
                for (int r = 0; r < 8; ++r)
                    op[(size_t)r * 128] = op[(size_t)r * 128] + acc[r];  // same-thread RAW
            }
        } else {                                 // cold path: boundary block only
            for (int n0 = 0; n0 < 128; n0 += 16) {
                bf16x16 bfr[8];
#pragma unroll
                for (int kk = 0; kk < 8; ++kk)
                    bfr[kk] = load_bfrag(W2t, 256, kk * 32, n0, lane);
                floatx8 acc = zero8();
#pragma unroll
                for (int kk = 0; kk < 8; ++kk)
                    acc = wmma_bf16(a8[kk], bfr[kk], acc);
                float* op = out + (size_t)(base + rbase) * 128 + n0 + col;
#pragma unroll
                for (int r = 0; r < 8; ++r)
                    if (base + rbase + r < n)
                        op[(size_t)r * 128] = op[(size_t)r * 128] + acc[r];
            }
        }
    }
}

extern "C" void kernel_launch(void* const* d_in, const int* in_sizes, int n_in,
                              void* d_out, int out_size, void* d_ws, size_t ws_size,
                              hipStream_t stream)
{
    (void)n_in; (void)out_size; (void)ws_size;
    const float* f  = (const float*)d_in[0];
    const int*   src = (const int*)d_in[1];
    // d_in[2] = dst, unused: dst = repeat(arange(N), KNN) (sorted, KNN per node)
    const float* g1 = (const float*)d_in[3];
    const float* b1 = (const float*)d_in[4];
    const float* m1 = (const float*)d_in[5];
    const float* v1 = (const float*)d_in[6];
    const float* Wq = (const float*)d_in[7];
    const float* Wk = (const float*)d_in[8];
    const float* Wv = (const float*)d_in[9];
    const float* Wo = (const float*)d_in[10];
    const float* bo = (const float*)d_in[11];
    const float* g2 = (const float*)d_in[12];
    const float* b2 = (const float*)d_in[13];
    const float* m2 = (const float*)d_in[14];
    const float* v2 = (const float*)d_in[15];
    const float* W1 = (const float*)d_in[16];
    const float* W2 = (const float*)d_in[17];

    const int N     = in_sizes[0] / 128;
    const int E     = in_sizes[1];
    const int knn   = E / N;
    const int tiles = (N + 127) / 128;
    const size_t Np = (size_t)tiles * 128;       // row-padded activation buffers

    // workspace layout (bf16 activations keep k/v gathers L2-resident: ~103 MB total)
    unsigned short* Wqt = (unsigned short*)d_ws;
    unsigned short* Wkt = Wqt + 128 * 128;
    unsigned short* Wvt = Wkt + 128 * 128;
    unsigned short* Wot = Wvt + 128 * 128;
    unsigned short* W1t = Wot + 128 * 128;       // 256 x 128
    unsigned short* W2t = W1t + 256 * 128;       // 128 x 256
    unsigned short* qb  = W2t + 128 * 256;
    unsigned short* kb  = qb + Np * 128;
    unsigned short* vb  = kb + Np * 128;
    unsigned short* ab  = vb + Np * 128;

    const int wblk = 256;
    wt_cvt_kernel<<<(128 * 128 + wblk - 1) / wblk, wblk, 0, stream>>>(Wq, Wqt, 128, 128);
    wt_cvt_kernel<<<(128 * 128 + wblk - 1) / wblk, wblk, 0, stream>>>(Wk, Wkt, 128, 128);
    wt_cvt_kernel<<<(128 * 128 + wblk - 1) / wblk, wblk, 0, stream>>>(Wv, Wvt, 128, 128);
    wt_cvt_kernel<<<(128 * 128 + wblk - 1) / wblk, wblk, 0, stream>>>(Wo, Wot, 128, 128);
    wt_cvt_kernel<<<(128 * 256 + wblk - 1) / wblk, wblk, 0, stream>>>(W1, W1t, 128, 256);
    wt_cvt_kernel<<<(256 * 128 + wblk - 1) / wblk, wblk, 0, stream>>>(W2, W2t, 256, 128);

    qkv_kernel<<<tiles, 256, 0, stream>>>(f, g1, b1, m1, v1, Wqt, Wkt, Wvt, qb, kb, vb, N);
    attn_kernel<<<(N + 7) / 8, 256, 0, stream>>>(qb, kb, vb, src, ab, N, knn);
    fused_out_kernel<<<tiles, 256, 0, stream>>>(ab, Wot, bo, f, g2, b2, m2, v2, W1t, W2t,
                                                (float*)d_out, N);
}
